// GSSM_41609643164080
// MI455X (gfx1250) — matrix operand order repaired and checked by reading, not compiled, not run on hardware
//
#include <hip/hip_runtime.h>
#include <math.h>

typedef __attribute__((ext_vector_type(16))) _Float16 v16h;
typedef __attribute__((ext_vector_type(8)))  float    v8f;

#define B_SZ 32
#define T_SZ 2048
#define DIN  512
#define DM   1024

// workspace layout (bytes)
//  [0, 3MB)          : W_lam/W_delt/W_inp as f16 B-fragments (786432 dwords)
//  [3MB, +64KB)      : xsum (32 x 512 f32)
//  [.., +128KB)      : s    (32 x 1024 f32)
#define WSW_DWORDS (786432u)
#define XSUM_OFF   (WSW_DWORDS * 4u)
#define S_OFF      (XSUM_OFF + 32u * 512u * 4u)

// LDS partition (bytes) for k_fused
#define BBUF_DW     (192u * 256u)            // 3 mats * 4 mgroups * 16 ksteps frags
#define BBUF_BYTES  (BBUF_DW * 4u)           // 196608
#define XFRAG_OFF   BBUF_BYTES
#define XFRAG_STR   268u                     // dwords per x fragment (2x132 halves)
#define XFRAG_BYTES (64u * XFRAG_STR * 4u)   // 68608
#define OUTB_OFF    (XFRAG_OFF + XFRAG_BYTES)
#define OUTB_BYTES  (3u * 64u * 65u * 4u)    // 49920
#define LDS_TOTAL   (OUTB_OFF + OUTB_BYTES)  // 315136 <= 320KB

__device__ __forceinline__ float softplusf(float v) {
    return v > 20.f ? v : log1pf(__expf(v));
}

// ---------------------------------------------------------------------------
// Convert the three D_MODEL weight matrices (512x1024 f32, row-major [d][m])
// into f16 WMMA B-fragments, with each lane's 8 dwords stored as two
// contiguous 16-byte runs so they can be fetched with ds_load_b128:
//   gid = ((mat*64 + mgroup)*16 + kstep)*256 + h*128 + lane*4 + q   (v = 4h+q)
//   dword v of lane = halves (K = kstep*32 + 16*(lane/16) + 2v, 2v+1 ;
//                             N = mgroup*16 + lane%16)
// ---------------------------------------------------------------------------
__global__ void k_convert_w(const float* __restrict__ Wl,
                            const float* __restrict__ Wd,
                            const float* __restrict__ Wi,
                            uint32_t* __restrict__ wsW) {
    uint32_t gid  = blockIdx.x * 256u + threadIdx.x;
    uint32_t q    = gid & 3u;
    uint32_t lane = (gid >> 2) & 31u;
    uint32_t h    = (gid >> 7) & 1u;
    uint32_t v    = h * 4u + q;
    uint32_t ks   = (gid >> 8) & 15u;
    uint32_t mgA  = gid >> 12;              // 0..191
    uint32_t mat  = mgA >> 6;
    uint32_t mg   = mgA & 63u;
    const float* W = (mat == 0) ? Wl : (mat == 1) ? Wd : Wi;
    uint32_t k = ks * 32u + ((lane >> 4) << 4) + 2u * v;
    uint32_t m = mg * 16u + (lane & 15u);
    float w0 = W[k * DM + m];
    float w1 = W[(k + 1u) * DM + m];
    union { _Float16 hh[2]; uint32_t u; } p;
    p.hh[0] = (_Float16)w0;   // low half  = even K
    p.hh[1] = (_Float16)w1;   // high half = odd K
    wsW[gid] = p.u;
}

// ---------------------------------------------------------------------------
// xsum[b][d] = sum_t x[b][t][d]   (streams x once, coalesced)
// ---------------------------------------------------------------------------
__global__ void k_xsum(const float* __restrict__ x, float* __restrict__ xsum) {
    int bb = blockIdx.x;
    int d  = threadIdx.x;
    const float* xp = x + (size_t)bb * T_SZ * DIN + d;
    float acc = 0.f;
    for (int t = 0; t < T_SZ; ++t) acc += xp[(size_t)t * DIN];
    xsum[bb * DIN + d] = acc;
}

// ---------------------------------------------------------------------------
// parity head: ang = pi*(xsum @ W_theta + T*b_theta); logits = [cos,sin]@W_par+b_par
// ---------------------------------------------------------------------------
__global__ void k_parity(const float* __restrict__ xsum,
                         const float* __restrict__ Wth, const float* __restrict__ bth,
                         const float* __restrict__ Wpar, const float* __restrict__ bpar,
                         float* __restrict__ out) {
    __shared__ float g[32][128];
    int tid = threadIdx.x;
    for (int p = tid; p < 32 * 64; p += 256) {
        int bb = p >> 6, k = p & 63;
        const float* xs = xsum + bb * DIN;
        float acc = 0.f;
        for (int d = 0; d < DIN; ++d) acc += xs[d] * Wth[d * 64 + k];
        float ang = 3.14159265358979323846f * (acc + (float)T_SZ * bth[k]);
        g[bb][k]      = cosf(ang);
        g[bb][64 + k] = sinf(ang);
    }
    __syncthreads();
    if (tid < 64) {
        int bb = tid >> 1, j = tid & 1;
        float acc = bpar[j];
        for (int i = 0; i < 128; ++i) acc += g[bb][i] * Wpar[i * 2 + j];
        out[bb * 2 + j] = acc;
    }
}

// ---------------------------------------------------------------------------
// Fused triple-GEMM (f16 WMMA) + sequential SSM scan.
// grid = (16 M-tiles of 64, 32 batches); block = 384 threads = 12 waves.
// Wave w: mat = w/4, tpair = (w>>1)&1, mgp = w&1 -> 2x2 register tile of
// 16x16 wmma accumulators: every A and B fragment fetched from LDS feeds two
// wmmas, and each fragment is read with two ds_load_b128 (2 ds instrs).
// B fragments (constant weights) live in LDS for the whole kernel.
// ---------------------------------------------------------------------------
__global__ void __launch_bounds__(384)
k_fused(const float* __restrict__ x, const uint32_t* __restrict__ wsW,
        const float* __restrict__ bias_lam, const float* __restrict__ bias_delt,
        const float* __restrict__ bias_inp, float* __restrict__ s_out) {
    extern __shared__ char smem[];
    uint32_t* Bb   = (uint32_t*)smem;                  // B fragments, 192 frags
    uint32_t* xf   = (uint32_t*)(smem + XFRAG_OFF);    // 64 frags * 268 dwords
    float*    outb = (float*)(smem + OUTB_OFF);        // [3][64][65] f32

    const int tid   = threadIdx.x;
    const int lane  = tid & 31;
    const int w     = tid >> 5;        // 0..11
    const int mat   = w >> 2;          // 0..2
    const int tpair = (w >> 1) & 1;    // 0..1
    const int mgp   = w & 1;           // 0..1
    const int bb    = blockIdx.y;
    const int mbase = blockIdx.x * 64;

    // scan state (threads 0..63 own channel mloc)
    float s = 0.f;
    const int   mloc = tid & 63;
    const float c_bl = bias_lam[mbase + mloc];
    const float c_bd = bias_delt[mbase + mloc];
    const float c_bi = bias_inp[mbase + mloc];

    // ---- one-time: stage this tile's B fragments into LDS (identity copy) ----
    {
        const uint32_t mg0 = (uint32_t)blockIdx.x * 4u;
        for (uint32_t idx = tid; idx < BBUF_DW; idx += 384u) {
            uint32_t d   = idx & 4095u;          // within one mgroup (16 frags)
            uint32_t lmg = (idx >> 12) & 3u;
            uint32_t mm  = idx >> 14;            // mat
            Bb[idx] = wsW[((mm * 64u + mg0 + lmg) * 16u) * 256u + d];
        }
    }
    __syncthreads();

    const float* xb = x + (size_t)bb * T_SZ * DIN;
    const int T0  = 2 * tpair;   // first of two t-tiles
    const int M0l = 2 * mgp;     // first of two local m-groups

    for (int c = 0; c < T_SZ / 64; ++c) {
        // ---- stage x chunk into A-fragment LDS layout (f32 -> f16) ----
        const float* xc = xb + (size_t)c * 64 * DIN;
        for (int idx = tid; idx < 64 * 256; idx += 384) {
            int r  = idx >> 8;        // t row in chunk
            int kp = idx & 255;       // k-pair
            float2 xv = *(const float2*)(xc + r * DIN + 2 * kp);
            union { _Float16 hh[2]; uint32_t u; } p;
            p.hh[0] = (_Float16)xv.x;
            p.hh[1] = (_Float16)xv.y;
            int ksf = kp >> 4, kk2 = kp & 15;
            int ln  = (r & 15) + ((kk2 >> 3) << 4);
            int vv  = kk2 & 7;
            int f   = ((r >> 4) << 4) + ksf;
            xf[f * XFRAG_STR + (vv >> 2) * 132 + (ln << 2) + (vv & 3)] = p.u;
            if (c + 1 < T_SZ / 64)
                __builtin_prefetch(xc + 64 * DIN + r * DIN + 2 * kp, 0, 0);
        }
        __syncthreads();

        // ---- WMMA GEMM: 2x2 register tile per wave, b128 operand fetch ----
        v8f zero = {};
        v8f a00 = zero, a01 = zero, a10 = zero, a11 = zero;

        for (int ks = 0; ks < 16; ++ks) {
            union { v16h h; uint4 q[2]; } A0, A1, B0, B1;
            const uint32_t* ab0 = xf + (T0 * 16 + ks) * XFRAG_STR;
            const uint32_t* ab1 = xf + ((T0 + 1) * 16 + ks) * XFRAG_STR;
            const uint32_t* bb0 = Bb + ((mat * 4 + M0l) * 16 + ks) * 256;
            const uint32_t* bb1 = Bb + ((mat * 4 + M0l + 1) * 16 + ks) * 256;
            A0.q[0] = *(const uint4*)(ab0 + lane * 4);
            A0.q[1] = *(const uint4*)(ab0 + 132 + lane * 4);
            A1.q[0] = *(const uint4*)(ab1 + lane * 4);
            A1.q[1] = *(const uint4*)(ab1 + 132 + lane * 4);
            B0.q[0] = *(const uint4*)(bb0 + lane * 4);
            B0.q[1] = *(const uint4*)(bb0 + 128 + lane * 4);
            B1.q[0] = *(const uint4*)(bb1 + lane * 4);
            B1.q[1] = *(const uint4*)(bb1 + 128 + lane * 4);
            a00 = __builtin_amdgcn_wmma_f32_16x16x32_f16(false, A0.h, false, B0.h, (short)0, a00, false, false);
            a01 = __builtin_amdgcn_wmma_f32_16x16x32_f16(false, A0.h, false, B1.h, (short)0, a01, false, false);
            a10 = __builtin_amdgcn_wmma_f32_16x16x32_f16(false, A1.h, false, B0.h, (short)0, a10, false, false);
            a11 = __builtin_amdgcn_wmma_f32_16x16x32_f16(false, A1.h, false, B1.h, (short)0, a11, false, false);
        }

        // ---- spill chunk pre-activations to LDS (C row = v + 8*(lane>=16)) ----
        {
            int mc    = lane & 15;
            int rbase = (lane >> 4) << 3;
            #pragma unroll
            for (int vv = 0; vv < 8; ++vv) {
                int r0 = T0 * 16 + rbase + vv;
                int r1 = (T0 + 1) * 16 + rbase + vv;
                outb[(mat * 64 + r0) * 65 + M0l * 16 + mc]       = a00[vv];
                outb[(mat * 64 + r0) * 65 + (M0l + 1) * 16 + mc] = a01[vv];
                outb[(mat * 64 + r1) * 65 + M0l * 16 + mc]       = a10[vv];
                outb[(mat * 64 + r1) * 65 + (M0l + 1) * 16 + mc] = a11[vv];
            }
        }
        __syncthreads();

        // ---- sequential scan over this chunk's 64 timesteps ----
        if (tid < 64) {
            for (int t = 0; t < 64; ++t) {
                float pl = outb[(0 * 64 + t) * 65 + mloc] + c_bl;
                float pd = outb[(1 * 64 + t) * 65 + mloc] + c_bd;
                float pi = outb[(2 * 64 + t) * 65 + mloc] + c_bi;
                float lam = softplusf(pl);
                float del = softplusf(pd);
                float alpha = __expf(-del * lam);
                s = alpha * s + del * pi;
            }
        }
        __syncthreads();
    }
    if (tid < 64) s_out[bb * DM + mbase + mloc] = s;
}

// ---------------------------------------------------------------------------
// add_pred[b] = s[b] . W_add + b_add
// ---------------------------------------------------------------------------
__global__ void k_addpred(const float* __restrict__ s, const float* __restrict__ Wadd,
                          const float* __restrict__ badd, float* __restrict__ out) {
    __shared__ float red[256];
    int bb = blockIdx.x, tid = threadIdx.x;
    float acc = 0.f;
    for (int i = tid; i < DM; i += 256) acc += s[bb * DM + i] * Wadd[i];
    red[tid] = acc;
    __syncthreads();
    for (int o = 128; o > 0; o >>= 1) {
        if (tid < o) red[tid] += red[tid + o];
        __syncthreads();
    }
    if (tid == 0) out[64 + bb] = red[0] + badd[0];
}

extern "C" void kernel_launch(void* const* d_in, const int* in_sizes, int n_in,
                              void* d_out, int out_size, void* d_ws, size_t ws_size,
                              hipStream_t stream) {
    const float* x   = (const float*)d_in[0];
    const float* Wth = (const float*)d_in[1];
    const float* bth = (const float*)d_in[2];
    const float* Wl  = (const float*)d_in[3];
    const float* bl  = (const float*)d_in[4];
    const float* Wd  = (const float*)d_in[5];
    const float* bd  = (const float*)d_in[6];
    const float* Wi  = (const float*)d_in[7];
    const float* bi  = (const float*)d_in[8];
    const float* Wp  = (const float*)d_in[9];
    const float* bp  = (const float*)d_in[10];
    const float* Wa  = (const float*)d_in[11];
    const float* ba  = (const float*)d_in[12];
    float* out = (float*)d_out;

    uint32_t* wsW  = (uint32_t*)d_ws;
    float*    xsum = (float*)((char*)d_ws + XSUM_OFF);
    float*    sbuf = (float*)((char*)d_ws + S_OFF);

    k_convert_w<<<WSW_DWORDS / 256, 256, 0, stream>>>(Wl, Wd, Wi, wsW);
    k_xsum<<<32, 512, 0, stream>>>(x, xsum);
    k_parity<<<1, 256, 0, stream>>>(xsum, Wth, bth, Wp, bp, out);
    k_fused<<<dim3(16, 32), 384, LDS_TOTAL, stream>>>(x, wsW, bl, bd, bi, sbuf);
    k_addpred<<<32, 256, 0, stream>>>(sbuf, Wa, ba, out);
}
